// MergeAdapter_18416819765629
// MI455X (gfx1250) — matrix operand order repaired and checked by reading, not compiled, not run on hardware
//
#include <hip/hip_runtime.h>
#include <hip/hip_bf16.h>

// Problem constants (from reference)
#define N_ 16
#define S_ 2048
#define H_ 1024
#define K_ 8
#define D_ 256

#define BM 64           // S-rows per workgroup
#define MID_STRIDE 272  // D_ + 16 pad (shorts), keeps 16B alignment (272*2 = 544 = 34*16)

typedef __attribute__((ext_vector_type(16))) __bf16         v16bf;
typedef __attribute__((ext_vector_type(8)))  float          v8f;
typedef __attribute__((ext_vector_type(16))) unsigned short us16;
typedef __attribute__((ext_vector_type(8)))  unsigned short us8;
typedef __attribute__((ext_vector_type(4)))  float          f32x4;

__device__ __forceinline__ unsigned short f2bf(float f) {
    unsigned int u = __float_as_uint(f);
    u += 0x7FFFu + ((u >> 16) & 1u);   // round-to-nearest-even
    return (unsigned short)(u >> 16);
}

// ---- Stage 1a: merged_weight[n] = sum_k prob[n,k] * w[k], output bf16 ----
__global__ __launch_bounds__(256) void merge_w_kernel(
    const float* __restrict__ prob, const float* __restrict__ w,
    unsigned short* __restrict__ out, int per_n)
{
    int idx = blockIdx.x * 256 + threadIdx.x;      // grid sized exactly: N_*per_n
    int n   = idx / per_n;
    int rem = idx - n * per_n;
    float acc = 0.f;
#pragma unroll
    for (int k = 0; k < K_; ++k)
        acc += prob[n * K_ + k] * w[(size_t)k * per_n + rem];
    out[idx] = f2bf(acc);
}

// ---- Stage 1b: merged bias, f32 ----
__global__ __launch_bounds__(256) void merge_b_kernel(
    const float* __restrict__ prob, const float* __restrict__ b,
    float* __restrict__ out, int per_n)
{
    int idx = blockIdx.x * 256 + threadIdx.x;      // grid sized exactly: N_*per_n
    int n   = idx / per_n;
    int rem = idx - n * per_n;
    float acc = 0.f;
#pragma unroll
    for (int k = 0; k < K_; ++k)
        acc += prob[n * K_ + k] * b[k * per_n + rem];
    out[idx] = acc;
}

// ---- Stage 1c: convert activations to bf16 once (removes VALU cvt from hot loop) ----
__global__ __launch_bounds__(256) void x2bf_kernel(
    const float* __restrict__ x, unsigned short* __restrict__ xb)
{
    size_t idx = ((size_t)blockIdx.x * 256 + threadIdx.x) * 8;  // grid: N*S*H/8/256
    f32x4 v0 = *(const f32x4*)(x + idx);
    f32x4 v1 = *(const f32x4*)(x + idx + 4);
    us8 o;
#pragma unroll
    for (int i = 0; i < 4; ++i) { o[i] = f2bf(v0[i]); o[4 + i] = f2bf(v1[i]); }
    *(us8*)(xb + idx) = o;
}

// ---- Stage 2: fused adapter: out = x + (relu(x@Wd^T + bd)) @ Wu^T + bu ----
// Block: 256 threads = 8 waves (wave32). One block = 64 S-rows of one instance n.
__global__ __launch_bounds__(256) void adapter_kernel(
    const float* __restrict__ x,            // (N,S,H) f32 (residual)
    const unsigned short* __restrict__ xbf, // (N,S,H) bf16 (GEMM operand)
    const unsigned short* __restrict__ wdm, // (N,D,H) bf16 merged
    const float* __restrict__ bdm,          // (N,D)
    const unsigned short* __restrict__ wum, // (N,H,D) bf16 merged
    const float* __restrict__ bum,          // (N,H)
    float* __restrict__ out)                // (N,S,H) f32
{
    __shared__ unsigned short mid[BM * MID_STRIDE];  // relu'd bottleneck, bf16

    const int n    = blockIdx.y;
    const int row0 = blockIdx.x * BM;
    const int lane = threadIdx.x & 31;
    const int wave = threadIdx.x >> 5;
    const int ln   = lane & 15;   // tile column (N index) / A-row (M index)
    const int half = lane >> 4;   // K-half selector per WMMA lane layout

    // ================= Phase A: mid = relu(x_tile @ Wd^T + bd) =================
    {
        const int rt  = wave >> 1;        // row tile 0..3 (two waves share one)
        const int dt0 = (wave & 1) * 8;   // this wave's 8 d-tiles
        const unsigned short* wbase = wdm + (size_t)n * D_ * H_;

        v8f acc[8];
#pragma unroll
        for (int j = 0; j < 8; ++j) {
            float b = bdm[n * D_ + (dt0 + j) * 16 + ln];
            acc[j] = (v8f){b, b, b, b, b, b, b, b};
        }

        // A row for this lane: M = ln within the row tile, bf16 source
        const unsigned short* arow =
            xbf + ((size_t)n * S_ + row0 + rt * 16 + ln) * H_;

        for (int kk = 0; kk < H_; kk += 32) {
            // A fragment (16x32 bf16): lanes 0-15 hold K{0..15}, lanes 16-31 K{16..31}
            us8 a0 = *(const us8*)(arow + kk + half * 16);        // global_load_b128
            us8 a1 = *(const us8*)(arow + kk + 8 + half * 16);   // global_load_b128
            us16 au;
#pragma unroll
            for (int i = 0; i < 8; ++i) { au[i] = a0[i]; au[8 + i] = a1[i]; }
            v16bf a = __builtin_bit_cast(v16bf, au);
#pragma unroll
            for (int j = 0; j < 8; ++j) {
                // B fragment (32x16): lane col = ln -> Wd row d; 16 contiguous bf16 at K half
                const unsigned short* bp =
                    wbase + (size_t)((dt0 + j) * 16 + ln) * H_ + kk + half * 16;
                v16bf b = __builtin_bit_cast(v16bf, *(const us16*)bp);
                acc[j] = __builtin_amdgcn_wmma_f32_16x16x32_bf16(
                    false, a, false, b, (short)0, acc[j], false, false);
            }
        }

        // ReLU + store bf16 to LDS. C layout: c[r] = C[r + 8*half][ln]
#pragma unroll
        for (int j = 0; j < 8; ++j) {
            int d = (dt0 + j) * 16 + ln;
#pragma unroll
            for (int r = 0; r < 8; ++r) {
                int row = rt * 16 + r + half * 8;
                float v = acc[j][r];
                v = v > 0.f ? v : 0.f;
                mid[row * MID_STRIDE + d] = f2bf(v);
            }
        }
    }
    __syncthreads();

    // ================= Phase B: out = x + mid @ Wu^T + bu =================
    {
        const int rt    = wave >> 1;
        const int hbase = (wave & 1) * 32;   // 32 h-tiles per wave, 4 groups of 8
        const unsigned short* wbase = wum + (size_t)n * H_ * D_;
        const unsigned short* amid  = &mid[(size_t)(rt * 16 + ln) * MID_STRIDE];

        for (int g = 0; g < 4; ++g) {
            const int ht0 = hbase + g * 8;
            v8f acc[8];
#pragma unroll
            for (int j = 0; j < 8; ++j) {
                float b = bum[n * H_ + (ht0 + j) * 16 + ln];
                acc[j] = (v8f){b, b, b, b, b, b, b, b};
            }
#pragma unroll
            for (int kk = 0; kk < D_; kk += 32) {
                us8 a0 = *(const us8*)(amid + kk + half * 16);       // ds_load_b128
                us8 a1 = *(const us8*)(amid + kk + 8 + half * 16);  // ds_load_b128
                us16 au;
#pragma unroll
                for (int i = 0; i < 8; ++i) { au[i] = a0[i]; au[8 + i] = a1[i]; }
                v16bf a = __builtin_bit_cast(v16bf, au);
#pragma unroll
                for (int j = 0; j < 8; ++j) {
                    const unsigned short* bp =
                        wbase + (size_t)((ht0 + j) * 16 + ln) * D_ + kk + half * 16;
                    v16bf b = __builtin_bit_cast(v16bf, *(const us16*)bp);
                    acc[j] = __builtin_amdgcn_wmma_f32_16x16x32_bf16(
                        false, a, false, b, (short)0, acc[j], false, false);
                }
            }
            // residual add + store
#pragma unroll
            for (int j = 0; j < 8; ++j) {
                int h = (ht0 + j) * 16 + ln;
#pragma unroll
                for (int r = 0; r < 8; ++r) {
                    int row = rt * 16 + r + half * 8;
                    size_t off = ((size_t)n * S_ + row0 + row) * H_ + h;
                    out[off] = x[off] + acc[j][r];
                }
            }
        }
    }
}

extern "C" void kernel_launch(void* const* d_in, const int* in_sizes, int n_in,
                              void* d_out, int out_size, void* d_ws, size_t ws_size,
                              hipStream_t stream) {
    (void)in_sizes; (void)n_in; (void)out_size; (void)ws_size;

    const float* x      = (const float*)d_in[0];  // (N,S,H)
    const float* prob   = (const float*)d_in[1];  // (N,K)
    const float* w_down = (const float*)d_in[2];  // (K,D,H)
    const float* b_down = (const float*)d_in[3];  // (K,D)
    const float* w_up   = (const float*)d_in[4];  // (K,H,D)
    const float* b_up   = (const float*)d_in[5];  // (K,H)
    float* out = (float*)d_out;

    // Workspace layout (bytes):
    //   [0)         : xbf bf16, N*S*H*2 = 67108864
    //   [+0x4000000): wdm bf16, N*D*H*2 = 8388608
    //   [... )      : wum bf16, N*H*D*2 = 8388608
    //   [... )      : bdm f32,  N*D*4   = 16384
    //   [... )      : bum f32,  N*H*4   = 65536   (total ~83.9 MB)
    char* ws = (char*)d_ws;
    const size_t xb_bytes = (size_t)N_ * S_ * H_ * 2;
    const size_t w_bytes  = (size_t)N_ * D_ * H_ * 2;
    unsigned short* xbf = (unsigned short*)ws;
    unsigned short* wdm = (unsigned short*)(ws + xb_bytes);
    unsigned short* wum = (unsigned short*)(ws + xb_bytes + w_bytes);
    float* bdm = (float*)(ws + xb_bytes + 2 * w_bytes);
    float* bum = (float*)(ws + xb_bytes + 2 * w_bytes + (size_t)N_ * D_ * 4);

    // Stage 1: merge expert weights/biases per instance; convert x to bf16
    merge_w_kernel<<<(N_ * D_ * H_) / 256, 256, 0, stream>>>(prob, w_down, wdm, D_ * H_);
    merge_w_kernel<<<(N_ * H_ * D_) / 256, 256, 0, stream>>>(prob, w_up,   wum, H_ * D_);
    merge_b_kernel<<<(N_ * D_) / 256, 256, 0, stream>>>(prob, b_down, bdm, D_);
    merge_b_kernel<<<(N_ * H_) / 256, 256, 0, stream>>>(prob, b_up,   bum, H_);
    x2bf_kernel<<<((size_t)N_ * S_ * H_) / 8 / 256, 256, 0, stream>>>(x, xbf);

    // Stage 2: fused down->relu->up->residual with WMMA bf16
    dim3 grid(S_ / BM, N_);
    adapter_kernel<<<grid, 256, 0, stream>>>(x, xbf, wdm, bdm, wum, bum, out);
}
